// PointProcessNeuralODE_62491774157228
// MI455X (gfx1250) — matrix operand order repaired and checked by reading, not compiled, not run on hardware
//
#include <hip/hip_runtime.h>
#include <hip/hip_bf16.h>
#include <math.h>

// ---------- types ----------
typedef __attribute__((ext_vector_type(16))) __bf16 bf16x16;
typedef __attribute__((ext_vector_type(8)))  float  f32x8;

union FragU {
  bf16x16 v;
  unsigned int u[8];
  unsigned short s[16];
};

__device__ __forceinline__ unsigned short f2bf(float x) {
  unsigned int u = __float_as_uint(x);
  u += 0x7fffu + ((u >> 16) & 1u);          // round-to-nearest-even
  return (unsigned short)(u >> 16);
}

__device__ __forceinline__ unsigned long long pack4bf(float4 v) {
  return  (unsigned long long)f2bf(v.x)
       | ((unsigned long long)f2bf(v.y) << 16)
       | ((unsigned long long)f2bf(v.z) << 32)
       | ((unsigned long long)f2bf(v.w) << 48);
}

__device__ __forceinline__ f32x8 zero8() {
  return (f32x8){0.f,0.f,0.f,0.f,0.f,0.f,0.f,0.f};
}

// A fragment (16x32 bf16) from LDS, row-major [rows][stride] of bf16 bits.
// Lane layout (ISA 7.12.2): m = lane&15; half=lane>>4; elems 0-7 -> K=half*8+j,
// elems 8-15 -> K=16+half*8+(j-8). Pairs are contiguous -> b128 loads.
__device__ __forceinline__ bf16x16 frag_a_lds(const unsigned short* base, int row, int stride) {
  const int lane = threadIdx.x & 31;
  const int half = lane >> 4;
  const unsigned short* p = base + row * stride;
  FragU f;
#pragma unroll
  for (int jj = 0; jj < 4; ++jj) {
    f.u[jj]     = *(const unsigned int*)(p + half * 8 + 2 * jj);
    f.u[4 + jj] = *(const unsigned int*)(p + 16 + half * 8 + 2 * jj);
  }
  return f.v;
}

// B fragment (32x16 bf16) from LDS stored K-transposed: baseT[n][k], stride elems.
// Lane layout: n = lane&15; elems 0-7 -> K=half*8+j, elems 8-15 -> K=16+half*8+(j-8).
__device__ __forceinline__ bf16x16 frag_b_lds(const unsigned short* baseT, int stride) {
  const int lane = threadIdx.x & 31;
  const int n = lane & 15, half = lane >> 4;
  const unsigned short* p = baseT + n * stride + half * 8;
  FragU f;
#pragma unroll
  for (int jj = 0; jj < 4; ++jj) {
    f.u[jj]     = *(const unsigned int*)(p + 2 * jj);
    f.u[4 + jj] = *(const unsigned int*)(p + 16 + 2 * jj);
  }
  return f.v;
}

__device__ __forceinline__ f32x8 wmma_bf16(bf16x16 a, bf16x16 b, f32x8 c) {
  return __builtin_amdgcn_wmma_f32_16x16x32_bf16(false, a, false, b, (short)0, c, false, false);
}

// ---------- generic GEMM: C[M,N] = act(A[M,K] @ W[K,N] + bias) ----------
// block = 256 threads (8 waves), tile 128x64, BK=32. act: 0=none, 1=relu.
// Software-pipelined: next K-tile's float4 global loads issue during WMMA.
__global__ __launch_bounds__(256) void gemm_bias_act(
    const float* __restrict__ A, const float* __restrict__ W,
    const float* __restrict__ bias, float* __restrict__ C,
    int M, int N, int K, int act)
{
  __shared__ __align__(16) unsigned short sA[128 * 32];  // [row][k]
  __shared__ __align__(16) unsigned short sB[64 * 32];   // transposed [n][k]
  const int tid  = threadIdx.x;
  const int wave = tid >> 5, lane = tid & 31;
  const int half = lane >> 4;
  const int m0 = blockIdx.x * 128;
  const int n0 = blockIdx.y * 64;

  // A-tile: 128x32 f32 = 1024 float4 -> 4/thread; B-tile: 32x64 = 512 float4 -> 2/thread
  int arr[4], ac4[4];
#pragma unroll
  for (int i = 0; i < 4; ++i) { int idx4 = tid + i * 256; arr[i] = idx4 >> 3; ac4[i] = (idx4 & 7) * 4; }
  int brr[2], bc4[2];
#pragma unroll
  for (int i = 0; i < 2; ++i) { int idx4 = tid + i * 256; brr[i] = idx4 >> 4; bc4[i] = (idx4 & 15) * 4; }

  float4 ra[4], rb[2];
  auto load_tiles = [&](int k0) {
#pragma unroll
    for (int i = 0; i < 4; ++i)
      ra[i] = *(const float4*)(A + (size_t)(m0 + arr[i]) * K + (k0 + ac4[i]));
#pragma unroll
    for (int i = 0; i < 2; ++i)
      rb[i] = *(const float4*)(W + (size_t)(k0 + brr[i]) * N + (n0 + bc4[i]));
  };

  f32x8 acc[4];
#pragma unroll
  for (int t = 0; t < 4; ++t) acc[t] = zero8();

  load_tiles(0);
  for (int k0 = 0; k0 < K; k0 += 32) {
#pragma unroll
    for (int i = 0; i < 4; ++i)
      *(unsigned long long*)(sA + arr[i] * 32 + ac4[i]) = pack4bf(ra[i]);
#pragma unroll
    for (int i = 0; i < 2; ++i) {
      sB[(bc4[i] + 0) * 32 + brr[i]] = f2bf(rb[i].x);
      sB[(bc4[i] + 1) * 32 + brr[i]] = f2bf(rb[i].y);
      sB[(bc4[i] + 2) * 32 + brr[i]] = f2bf(rb[i].z);
      sB[(bc4[i] + 3) * 32 + brr[i]] = f2bf(rb[i].w);
    }
    __syncthreads();
    if (k0 + 32 < K) load_tiles(k0 + 32);   // uniform branch; loads fly over WMMAs

    // Load all fragments first so the scheduler can use partial dscnt waits.
    bf16x16 af = frag_a_lds(sA, wave * 16 + (lane & 15), 32);
    bf16x16 bfm[4];
#pragma unroll
    for (int t = 0; t < 4; ++t) bfm[t] = frag_b_lds(sB + t * 16 * 32, 32);
#pragma unroll
    for (int t = 0; t < 4; ++t) acc[t] = wmma_bf16(af, bfm[t], acc[t]);
    __syncthreads();
  }

#pragma unroll
  for (int t = 0; t < 4; ++t) {
#pragma unroll
    for (int r = 0; r < 8; ++r) {
      int m = m0 + wave * 16 + r + 8 * half;
      int n = n0 + t * 16 + (lane & 15);
      float v = acc[t][r] + bias[n];
      if (act == 1) v = fmaxf(v, 0.f);
      C[(size_t)m * N + n] = v;
    }
  }
}

// ---------- flash attention ----------
// qkv: [B,S,3D], Out: [B,S,D]. Block = 128 threads (4 waves), 64 q-rows/block,
// 16 q-rows/wave, 32 keys per step, HD=32 -> score K-dim = exactly one WMMA.
// Softmax runs in the exp2 domain (log2e folded into the Q scale).
__global__ __launch_bounds__(128) void flash_attn(
    const float* __restrict__ qkv, float* __restrict__ Out,
    int S, int Dm, float scale_log2e)
{
  __shared__ __align__(16) unsigned short sK[32 * 32];     // [key][kd]
  __shared__ __align__(16) unsigned short sVt[32 * 32];    // [d][key]
  __shared__ __align__(16) unsigned short sP[4 * 16 * 32]; // per-wave [q][key]
  const int tid  = threadIdx.x;
  const int wave = tid >> 5, lane = tid & 31;
  const int m = lane & 15, half = lane >> 4;
  const int q0 = blockIdx.x * 64;
  const int hi = blockIdx.y;
  const int bi = blockIdx.z;
  const int D3 = 3 * Dm;
  const int HDc = 32;

  FragU qf;  // Q fragment, scale * log2(e) folded in
  {
    const int srow = q0 + wave * 16 + m;
    const float* qp = qkv + (size_t)(bi * S + srow) * D3 + hi * HDc;
#pragma unroll
    for (int j = 0; j < 16; ++j) {
      int kd = (j < 8) ? (half * 8 + j) : (16 + half * 8 + (j - 8));
      qf.s[j] = f2bf(qp[kd] * scale_log2e);
    }
  }

  f32x8 o0 = zero8(), o1 = zero8();
  float run_max[8], run_sum[8];
#pragma unroll
  for (int r = 0; r < 8; ++r) { run_max[r] = -1e30f; run_sum[r] = 0.f; }

  unsigned short* myP = sP + wave * (16 * 32);

  // K/V tiles: 32x32 f32 each = 256 float4 each -> 2/thread each
  int kkey[2], kc4[2];
#pragma unroll
  for (int i = 0; i < 2; ++i) { int idx4 = tid + i * 128; kkey[i] = idx4 >> 3; kc4[i] = (idx4 & 7) * 4; }
  float4 rk[2], rv[2];
  auto load_kv = [&](int kb) {
#pragma unroll
    for (int i = 0; i < 2; ++i) {
      const float* base = qkv + (size_t)(bi * S + kb + kkey[i]) * D3 + hi * HDc;
      rk[i] = *(const float4*)(base + Dm + kc4[i]);
      rv[i] = *(const float4*)(base + 2 * Dm + kc4[i]);
    }
  };

  load_kv(0);
  for (int kb = 0; kb < S; kb += 32) {
#pragma unroll
    for (int i = 0; i < 2; ++i) {            // K packed b64; V transposed b16 scatter
      *(unsigned long long*)(sK + kkey[i] * 32 + kc4[i]) = pack4bf(rk[i]);
      sVt[(kc4[i] + 0) * 32 + kkey[i]] = f2bf(rv[i].x);
      sVt[(kc4[i] + 1) * 32 + kkey[i]] = f2bf(rv[i].y);
      sVt[(kc4[i] + 2) * 32 + kkey[i]] = f2bf(rv[i].z);
      sVt[(kc4[i] + 3) * 32 + kkey[i]] = f2bf(rv[i].w);
    }
    __syncthreads();
    if (kb + 32 < S) load_kv(kb + 32);       // pipeline next tile over the math

    // All LDS fragment loads issued up front: K frags feed the score WMMAs,
    // V frags' latency hides under the softmax VALU below.
    bf16x16 kb0 = frag_b_lds(sK, 32);                 // keys kb..kb+15
    bf16x16 kb1 = frag_b_lds(sK + 16 * 32, 32);       // keys kb+16..kb+31
    bf16x16 vb0 = frag_b_lds(sVt, 32);
    bf16x16 vb1 = frag_b_lds(sVt + 16 * 32, 32);
    f32x8 c0 = wmma_bf16(qf.v, kb0, zero8());
    f32x8 c1 = wmma_bf16(qf.v, kb1, zero8());

    float p0[8], p1[8];
#pragma unroll
    for (int r = 0; r < 8; ++r) {                     // online softmax (base 2)
      float tm = fmaxf(c0[r], c1[r]);
#pragma unroll
      for (int off = 1; off < 16; off <<= 1) tm = fmaxf(tm, __shfl_xor(tm, off));
      float nm = fmaxf(run_max[r], tm);
      float alpha = exp2f(run_max[r] - nm);
      float e0 = exp2f(c0[r] - nm);
      float e1 = exp2f(c1[r] - nm);
      float ps = e0 + e1;
#pragma unroll
      for (int off = 1; off < 16; off <<= 1) ps += __shfl_xor(ps, off);
      run_sum[r] = run_sum[r] * alpha + ps;
      run_max[r] = nm;
      o0[r] *= alpha; o1[r] *= alpha;
      p0[r] = e0; p1[r] = e1;
    }

#pragma unroll
    for (int r = 0; r < 8; ++r) {                     // P tile -> wave-private LDS
      int row = r + 8 * half;
      myP[row * 32 + (lane & 15)]      = f2bf(p0[r]);
      myP[row * 32 + 16 + (lane & 15)] = f2bf(p1[r]);
    }
    asm volatile("s_wait_dscnt 0x0" ::: "memory");    // wave-internal LDS RAW

    bf16x16 pa = frag_a_lds(myP, m, 32);
    o0 = wmma_bf16(pa, vb0, o0);
    o1 = wmma_bf16(pa, vb1, o1);
    __syncthreads();
  }

#pragma unroll
  for (int r = 0; r < 8; ++r) {
    float inv = 1.f / run_sum[r];
    int srow = q0 + wave * 16 + r + 8 * half;
    float* op = Out + (size_t)(bi * S + srow) * Dm + hi * HDc;
    op[(lane & 15)]      = o0[r] * inv;
    op[16 + (lane & 15)] = o1[r] * inv;
  }
}

// ---------- out = LayerNorm(x + res) over D=256, one wave per row ----------
__global__ __launch_bounds__(256) void ln_residual(
    const float* __restrict__ x, const float* __restrict__ res,
    const float* __restrict__ gs, const float* __restrict__ gb,
    float* __restrict__ out)
{
  const int wave = threadIdx.x >> 5, lane = threadIdx.x & 31;
  const int row = blockIdx.x * 8 + wave;
  const float* xp = x   + (size_t)row * 256;
  const float* rp = res + (size_t)row * 256;
  float v[8]; float sum = 0.f;
#pragma unroll
  for (int i = 0; i < 8; ++i) { v[i] = xp[lane + 32 * i] + rp[lane + 32 * i]; sum += v[i]; }
#pragma unroll
  for (int off = 1; off < 32; off <<= 1) sum += __shfl_xor(sum, off);
  float mean = sum * (1.f / 256.f);
  float var = 0.f;
#pragma unroll
  for (int i = 0; i < 8; ++i) { float d = v[i] - mean; var += d * d; }
#pragma unroll
  for (int off = 1; off < 32; off <<= 1) var += __shfl_xor(var, off);
  float inv = rsqrtf(var * (1.f / 256.f) + 1e-5f);
  float* op = out + (size_t)row * 256;
#pragma unroll
  for (int i = 0; i < 8; ++i) {
    int c = lane + 32 * i;
    op[c] = (v[i] - mean) * inv * gs[c] + gb[c];
  }
}

// ---------- intensity = softplus(a_row . w + b), a: [M,256], one wave/row ----------
__global__ __launch_bounds__(256) void intensity_head(
    const float* __restrict__ a, const float* __restrict__ w,
    const float* __restrict__ b, float* __restrict__ out)
{
  const int wave = threadIdx.x >> 5, lane = threadIdx.x & 31;
  const int row = blockIdx.x * 8 + wave;
  const float* ap = a + (size_t)row * 256;
  float acc = 0.f;
#pragma unroll
  for (int i = 0; i < 8; ++i) acc += ap[lane + 32 * i] * w[lane + 32 * i];
#pragma unroll
  for (int off = 1; off < 32; off <<= 1) acc += __shfl_xor(acc, off);
  if (lane == 0) {
    float xx = acc + b[0];
    out[row] = (xx > 20.f) ? xx : log1pf(__expf(xx));
  }
}

// ---------- mark logits: out[M,13] = a[M,128] @ w[128,13] + b ----------
__global__ __launch_bounds__(256) void mark2_head(
    const float* __restrict__ a, const float* __restrict__ w,
    const float* __restrict__ b, float* __restrict__ out, int M)
{
  int idx = blockIdx.x * blockDim.x + threadIdx.x;
  int row = idx >> 4, e = idx & 15;
  if (row >= M || e >= 13) return;
  const float* ap = a + (size_t)row * 128;
  float acc = b[e];
#pragma unroll 4
  for (int k = 0; k < 128; ++k) acc += ap[k] * w[k * 13 + e];
  out[(size_t)row * 13 + e] = acc;
}

extern "C" void kernel_launch(void* const* d_in, const int* in_sizes, int n_in,
                              void* d_out, int out_size, void* d_ws, size_t ws_size,
                              hipStream_t stream)
{
  (void)in_sizes; (void)n_in; (void)out_size; (void)ws_size;
  const float* x_seq   = (const float*)d_in[0];
  /* d_in[1] inter_event_times: unused by the returned outputs */
  const float* emb_w   = (const float*)d_in[2];
  const float* emb_b   = (const float*)d_in[3];
  const float* qkv_w   = (const float*)d_in[4];
  const float* qkv_b   = (const float*)d_in[5];
  const float* out_w   = (const float*)d_in[6];
  const float* out_b   = (const float*)d_in[7];
  const float* ln1_s   = (const float*)d_in[8];
  const float* ln1_b   = (const float*)d_in[9];
  const float* ln2_s   = (const float*)d_in[10];
  const float* ln2_b   = (const float*)d_in[11];
  const float* ff1_w   = (const float*)d_in[12];
  const float* ff1_b   = (const float*)d_in[13];
  const float* ff2_w   = (const float*)d_in[14];
  const float* ff2_b   = (const float*)d_in[15];
  const float* int1_w  = (const float*)d_in[16];
  const float* int1_b  = (const float*)d_in[17];
  const float* int2_w  = (const float*)d_in[18];
  const float* int2_b  = (const float*)d_in[19];
  const float* mark1_w = (const float*)d_in[20];
  const float* mark1_b = (const float*)d_in[21];
  const float* mark2_w = (const float*)d_in[22];
  const float* mark2_b = (const float*)d_in[23];

  const int Bn = 4, S = 2048, D = 256, Hh = 8, FF = 1024, Ll = 4;
  const int M = Bn * S;  // 8192

  float* ws   = (float*)d_ws;
  float* hbuf = ws;                          // [M, D]
  float* bufB = hbuf + (size_t)M * D;        // [M, FF]  (qkv / ff1)
  float* bufC = bufB + (size_t)M * FF;       // [M, D]   (attn out / ff2 / int1)
  float* bufD = bufC + (size_t)M * D;        // [M, D]   (out-proj / mark1)

  float* out_int  = (float*)d_out;           // [M, 1]
  float* out_mark = out_int + M;             // [M, 13]
  float* out_h    = out_mark + (size_t)M * 13;  // [M, D]

  dim3 blk(256);
  const float scale_log2e = (1.f / sqrtf(32.f)) * 1.44269504088896340736f;

  // embedding: h = x @ emb_w + emb_b
  gemm_bias_act<<<dim3(M / 128, D / 64), blk, 0, stream>>>(
      x_seq, emb_w, emb_b, hbuf, M, D, 64, 0);

  for (int l = 0; l < Ll; ++l) {
    gemm_bias_act<<<dim3(M / 128, (3 * D) / 64), blk, 0, stream>>>(
        hbuf, qkv_w + (size_t)l * D * 3 * D, qkv_b + (size_t)l * 3 * D,
        bufB, M, 3 * D, D, 0);
    flash_attn<<<dim3(S / 64, Hh, Bn), dim3(128), 0, stream>>>(
        bufB, bufC, S, D, scale_log2e);
    gemm_bias_act<<<dim3(M / 128, D / 64), blk, 0, stream>>>(
        bufC, out_w + (size_t)l * D * D, out_b + (size_t)l * D, bufD, M, D, D, 0);
    ln_residual<<<dim3(M / 8), blk, 0, stream>>>(
        hbuf, bufD, ln1_s + (size_t)l * D, ln1_b + (size_t)l * D, hbuf);
    gemm_bias_act<<<dim3(M / 128, FF / 64), blk, 0, stream>>>(
        hbuf, ff1_w + (size_t)l * D * FF, ff1_b + (size_t)l * FF, bufB, M, FF, D, 1);
    gemm_bias_act<<<dim3(M / 128, D / 64), blk, 0, stream>>>(
        bufB, ff2_w + (size_t)l * FF * D, ff2_b + (size_t)l * D, bufC, M, D, FF, 0);
    float* lnout = (l == Ll - 1) ? out_h : hbuf;   // last LN writes h output slice
    ln_residual<<<dim3(M / 8), blk, 0, stream>>>(
        hbuf, bufC, ln2_s + (size_t)l * D, ln2_b + (size_t)l * D, lnout);
  }

  // heads (read final h from d_out slice)
  gemm_bias_act<<<dim3(M / 128, D / 64), blk, 0, stream>>>(
      out_h, int1_w, int1_b, bufC, M, D, D, 1);
  intensity_head<<<dim3(M / 8), blk, 0, stream>>>(bufC, int2_w, int2_b, out_int);
  gemm_bias_act<<<dim3(M / 128, 128 / 64), blk, 0, stream>>>(
      out_h, mark1_w, mark1_b, bufD, M, 128, D, 1);
  mark2_head<<<dim3((M * 16) / 256), blk, 0, stream>>>(
      bufD, mark2_w, mark2_b, out_mark, M);
}
